// PLIPGNN_50946902065763
// MI455X (gfx1250) — matrix operand order repaired and checked by reading, not compiled, not run on hardware
//
#include <hip/hip_runtime.h>
#include <hip/hip_bf16.h>

// ---------------------------------------------------------------------------
// PLIP-GNN (GINEConv x3 + GraphNorm) for MI455X / gfx1250.
// All dense layers run on v_wmma_f32_16x16x32_bf16 (f32 accumulate).
// Message passing (gather+relu+scatter) is fused with the edge-linear WMMA.
// GEMM: each wave owns two 16x16 col tiles sharing one A fragment (halves
// A-matrix traffic, issues 2 independent WMMAs per A load).
// ---------------------------------------------------------------------------

typedef __bf16 bf16_t;
typedef __bf16  v16bf __attribute__((ext_vector_type(16)));
typedef __bf16  v8bf  __attribute__((ext_vector_type(8)));
typedef float   v8f   __attribute__((ext_vector_type(8)));

#define NN_NODES 50000
#define NN_EDGES 800000
#define IN_DIM   64
#define HID      128
#define NGRAPH   128
#define NLAYER   3

// ---------------- workspace layout (bytes, all 256-aligned) ----------------
// bf16 weight region element offsets:
#define WB_LIG_W1  0
#define WB_LIG_W2  8192
#define WB_PROT_W1 24576
#define WB_PROT_W2 32768
#define WB_EDGE_W  49152      /* 3 x (32x128), K zero-padded 16->32 */
#define WB_NN_W1   61440
#define WB_NN_W2   110592
#define WB_FC_W1   159744
#define WB_ELEMS   176128

#define OFF_WB   ((size_t)0)
#define OFF_HF   ((size_t)352256)                 /* f32 h      [50000x128] */
#define OFF_TF   (OFF_HF + (size_t)25600000)      /* f32 tmp/aggr           */
#define OFF_HB   (OFF_TF + (size_t)25600000)      /* bf16 h                 */
#define OFF_TB   (OFF_HB + (size_t)12800000)      /* bf16 tmp               */
#define OFF_XB   (OFF_TB + (size_t)12800000)      /* bf16 x     [50000x64]  */
#define OFF_CNT  (OFF_XB + (size_t)6400000)       /* f32 [128]              */
#define OFF_MEAN (OFF_CNT + (size_t)512)          /* f32 [128x128]          */
#define OFF_VAR  (OFF_MEAN + (size_t)65536)       /* f32 [128x128]          */
#define OFF_G    (OFF_VAR + (size_t)65536)        /* f32 [128x128] pooled   */
#define OFF_GB   (OFF_G + (size_t)65536)          /* bf16 pooled            */
#define OFF_HEAD (OFF_GB + (size_t)32768)         /* f32 [128x128]          */

// -------------------------------- utilities --------------------------------
__global__ void k_zero_f32(float* p, int n) {
    int i = blockIdx.x * blockDim.x + threadIdx.x;
    if (i < n) p[i] = 0.0f;
}

__global__ void k_cvt_bf16(const float* __restrict__ s, bf16_t* __restrict__ d, int n) {
    int i = blockIdx.x * blockDim.x + threadIdx.x;
    if (i < n) d[i] = (bf16_t)s[i];
}

// edge_w [3][16][128] f32 -> [3][32][128] bf16, rows 16..31 zero
__global__ void k_cvt_edgew(const float* __restrict__ s, bf16_t* __restrict__ d) {
    int i = blockIdx.x * blockDim.x + threadIdx.x;   // over 3*32*128
    if (i >= NLAYER * 32 * HID) return;
    int col = i & (HID - 1);
    int k   = (i >> 7) & 31;
    int l   = i >> 12;
    d[i] = (k < 16) ? (bf16_t)s[l * 16 * HID + k * HID + col] : (bf16_t)0.0f;
}

// --------------------------- generic bf16 WMMA GEMM ------------------------
// C[M x 128] = act(A[M x K] @ B[K x 128] + bias); block = 128 thr (4 waves).
// Wave w computes col tiles at w*16 and w*16+64, sharing one A fragment.
// grid = (M/16).
__global__ void k_gemm_bf16(const bf16_t* __restrict__ A, int lda,
                            const bf16_t* __restrict__ B,
                            const float* __restrict__ bias,
                            float* __restrict__ outF,
                            bf16_t* __restrict__ outB,
                            int K, int relu) {
    int lane = threadIdx.x & 31;
    int wave = threadIdx.x >> 5;       // 0..3
    int row0 = blockIdx.x * 16;
    int r = lane & 15, half = lane >> 4;
    int col0 = wave * 16 + r;          // first col tile
    int col1 = col0 + 64;              // second col tile

    v8f acc0, acc1;
    float bv0 = bias[col0], bv1 = bias[col1];
#pragma unroll
    for (int i = 0; i < 8; ++i) { acc0[i] = bv0; acc1[i] = bv1; }

    const bf16_t* arow = A + (size_t)(row0 + r) * lda + half * 8;
    const bf16_t* brow = B + (size_t)(half * 8) * HID + col0;
    for (int k = 0; k < K; k += 32) {
        v8bf a0 = *(const v8bf*)(arow + k);
        v8bf a1 = *(const v8bf*)(arow + k + 16);
        v16bf a;
#pragma unroll
        for (int i = 0; i < 8; ++i) { a[i] = a0[i]; a[8 + i] = a1[i]; }
        v16bf b0, b1;
#pragma unroll
        for (int i = 0; i < 8; ++i) {
            b0[i] = brow[(size_t)(k + i) * HID];
            b0[8 + i] = brow[(size_t)(k + 16 + i) * HID];
            b1[i] = brow[(size_t)(k + i) * HID + 64];
            b1[8 + i] = brow[(size_t)(k + 16 + i) * HID + 64];
        }
        acc0 = __builtin_amdgcn_wmma_f32_16x16x32_bf16(false, a, false, b0,
                                                       (short)0, acc0, false, false);
        acc1 = __builtin_amdgcn_wmma_f32_16x16x32_bf16(false, a, false, b1,
                                                       (short)0, acc1, false, false);
    }
    size_t base0 = (size_t)(row0 + half * 8) * HID + col0;
#pragma unroll
    for (int v = 0; v < 8; ++v) {
        float o0 = acc0[v], o1 = acc1[v];
        if (relu) { o0 = fmaxf(o0, 0.0f); o1 = fmaxf(o1, 0.0f); }
        size_t i0 = base0 + (size_t)v * HID;
        if (outF) { outF[i0] = o0; outF[i0 + 64] = o1; }
        if (outB) { outB[i0] = (bf16_t)o0; outB[i0 + 64] = (bf16_t)o1; }
    }
}

// ------------- fused edge linear + gather + relu + scatter-add -------------
// block = 256 thr (8 waves); each block: 16 edges, each wave: 16 cols.
__global__ void k_edge_msg(const float* __restrict__ ea,
                           const int* __restrict__ src,
                           const int* __restrict__ dst,
                           const bf16_t* __restrict__ Bw,   // 32x128, padded
                           const float* __restrict__ eb,
                           const float* __restrict__ hf,
                           float* __restrict__ aggr) {
    int lane = threadIdx.x & 31;
    int wave = threadIdx.x >> 5;
    int e0 = blockIdx.x * 16;
    int r = lane & 15, half = lane >> 4;
    int col0 = wave * 16;
    int col = col0 + r;

    // A fragment: edge_attr tile [16 x 32], real K = 0..15, rest zero.
    v16bf a;
    const float* ap = ea + (size_t)(e0 + r) * 16 + half * 8;
#pragma unroll
    for (int i = 0; i < 8; ++i) { a[i] = (bf16_t)ap[i]; a[8 + i] = (bf16_t)0.0f; }

    v16bf b;
#pragma unroll
    for (int i = 0; i < 8; ++i) {
        b[i] = Bw[(size_t)(half * 8 + i) * HID + col];
        b[8 + i] = Bw[(size_t)(16 + half * 8 + i) * HID + col];
    }

    v8f acc;
    float bv = eb[col];
#pragma unroll
    for (int i = 0; i < 8; ++i) acc[i] = bv;
    acc = __builtin_amdgcn_wmma_f32_16x16x32_bf16(false, a, false, b,
                                                  (short)0, acc, false, false);
#pragma unroll
    for (int v = 0; v < 8; ++v) {
        int ei = e0 + half * 8 + v;
        int s = src[ei], d = dst[ei];
        float m = fmaxf(hf[(size_t)s * HID + col] + acc[v], 0.0f);
        unsafeAtomicAdd(&aggr[(size_t)d * HID + col], m);
    }
}

// ------------------------------- elementwise -------------------------------
__global__ void k_select_enc(const float* __restrict__ x,
                             const float* __restrict__ hlig,
                             float* __restrict__ hf, bf16_t* __restrict__ hb, int n) {
    int i = blockIdx.x * blockDim.x + threadIdx.x;
    if (i >= n) return;
    int row = i >> 7;
    bool p = x[(size_t)row * IN_DIM + (IN_DIM - 1)] > 0.5f;
    float v = p ? hf[i] : hlig[i];
    hf[i] = v; hb[i] = (bf16_t)v;
}

__global__ void k_add_cvt(float* __restrict__ hf, const float* __restrict__ tf,
                          bf16_t* __restrict__ hb, int n) {
    int i = blockIdx.x * blockDim.x + threadIdx.x;
    if (i >= n) return;
    float v = hf[i] + tf[i];
    hf[i] = v; hb[i] = (bf16_t)v;
}

__global__ void k_count(const int* __restrict__ batch, float* __restrict__ cnt, int n) {
    int i = blockIdx.x * blockDim.x + threadIdx.x;
    if (i < n) unsafeAtomicAdd(&cnt[batch[i]], 1.0f);
}

__global__ void k_gn_sum(const float* __restrict__ hf, const int* __restrict__ batch,
                         float* __restrict__ mean, int n) {
    int i = blockIdx.x * blockDim.x + threadIdx.x;
    if (i >= n) return;
    unsafeAtomicAdd(&mean[batch[i >> 7] * HID + (i & 127)], hf[i]);
}

__global__ void k_gn_meandiv(float* __restrict__ mean, const float* __restrict__ cnt, int n) {
    int i = blockIdx.x * blockDim.x + threadIdx.x;
    if (i < n) mean[i] /= fmaxf(cnt[i >> 7], 1.0f);
}

__global__ void k_gn_center(float* __restrict__ hf, const int* __restrict__ batch,
                            const float* __restrict__ mean, const float* __restrict__ ms,
                            float* __restrict__ var, int n) {
    int i = blockIdx.x * blockDim.x + threadIdx.x;
    if (i >= n) return;
    int c = i & 127, g = batch[i >> 7];
    float hc = hf[i] - mean[g * HID + c] * ms[c];
    hf[i] = hc;
    unsafeAtomicAdd(&var[g * HID + c], hc * hc);
}

__global__ void k_gn_norm(float* __restrict__ hf, bf16_t* __restrict__ hb,
                          const int* __restrict__ batch, const float* __restrict__ var,
                          const float* __restrict__ cnt, const float* __restrict__ w,
                          const float* __restrict__ b, int n) {
    int i = blockIdx.x * blockDim.x + threadIdx.x;
    if (i >= n) return;
    int c = i & 127, g = batch[i >> 7];
    float vv = var[g * HID + c] / fmaxf(cnt[g], 1.0f);
    float o = fmaxf(w[c] * hf[i] * rsqrtf(vv + 1e-5f) + b[c], 0.0f);
    hf[i] = o; hb[i] = (bf16_t)o;
}

__global__ void k_pool(const float* __restrict__ hf, const int* __restrict__ batch,
                       float* __restrict__ g, int n) {
    int i = blockIdx.x * blockDim.x + threadIdx.x;
    if (i >= n) return;
    unsafeAtomicAdd(&g[batch[i >> 7] * HID + (i & 127)], hf[i]);
}

__global__ void k_head(const float* __restrict__ hid, const float* __restrict__ w2,
                       const float* __restrict__ b2, float* __restrict__ out) {
    int g = threadIdx.x;  // one thread per graph (128)
    float s = 0.0f;
    for (int j = 0; j < HID; ++j) s += hid[g * HID + j] * w2[j];
    out[g] = s + b2[0];
}

// --------------------------------- driver ----------------------------------
extern "C" void kernel_launch(void* const* d_in, const int* in_sizes, int n_in,
                              void* d_out, int out_size, void* d_ws, size_t ws_size,
                              hipStream_t stream) {
    const float* x      = (const float*)d_in[0];
    const int*   ei     = (const int*)d_in[1];
    const int*   src    = ei;
    const int*   dst    = ei + NN_EDGES;
    const float* ea     = (const float*)d_in[2];
    const int*   batch  = (const int*)d_in[3];
    const float* lig_w1 = (const float*)d_in[4],  *lig_b1 = (const float*)d_in[5];
    const float* lig_w2 = (const float*)d_in[6],  *lig_b2 = (const float*)d_in[7];
    const float* prot_w1= (const float*)d_in[8],  *prot_b1= (const float*)d_in[9];
    const float* prot_w2= (const float*)d_in[10], *prot_b2= (const float*)d_in[11];
    const float* edge_w = (const float*)d_in[12], *edge_b = (const float*)d_in[13];
    const float* nn_w1  = (const float*)d_in[14], *nn_b1  = (const float*)d_in[15];
    const float* nn_w2  = (const float*)d_in[16], *nn_b2  = (const float*)d_in[17];
    const float* gn_w   = (const float*)d_in[18], *gn_b   = (const float*)d_in[19];
    const float* gn_ms  = (const float*)d_in[20];
    const float* fc_w1  = (const float*)d_in[21], *fc_b1  = (const float*)d_in[22];
    const float* fc_w2  = (const float*)d_in[23], *fc_b2  = (const float*)d_in[24];

    char* ws = (char*)d_ws;
    bf16_t* WB   = (bf16_t*)(ws + OFF_WB);
    float*  HF   = (float*)(ws + OFF_HF);
    float*  TF   = (float*)(ws + OFF_TF);
    bf16_t* HB   = (bf16_t*)(ws + OFF_HB);
    bf16_t* TB   = (bf16_t*)(ws + OFF_TB);
    bf16_t* XB   = (bf16_t*)(ws + OFF_XB);
    float*  CNT  = (float*)(ws + OFF_CNT);
    float*  MEAN = (float*)(ws + OFF_MEAN);
    float*  VAR  = (float*)(ws + OFF_VAR);
    float*  GP   = (float*)(ws + OFF_G);
    bf16_t* GB   = (bf16_t*)(ws + OFF_GB);
    float*  HEAD = (float*)(ws + OFF_HEAD);

    const int nEl = NN_NODES * HID;                 // 6.4M
    dim3 bEl(256), gEl((nEl + 255) / 256);
    dim3 gGemm(NN_NODES / 16), bGemm(128);

    auto cvt = [&](const float* s, bf16_t* d, int n) {
        k_cvt_bf16<<<(n + 255) / 256, 256, 0, stream>>>(s, d, n);
    };

    // --- weight / input conversion to bf16 ---
    cvt(lig_w1,  WB + WB_LIG_W1,  IN_DIM * HID);
    cvt(lig_w2,  WB + WB_LIG_W2,  HID * HID);
    cvt(prot_w1, WB + WB_PROT_W1, IN_DIM * HID);
    cvt(prot_w2, WB + WB_PROT_W2, HID * HID);
    k_cvt_edgew<<<(NLAYER * 32 * HID + 255) / 256, 256, 0, stream>>>(edge_w, WB + WB_EDGE_W);
    cvt(nn_w1, WB + WB_NN_W1, NLAYER * HID * HID);
    cvt(nn_w2, WB + WB_NN_W2, NLAYER * HID * HID);
    cvt(fc_w1, WB + WB_FC_W1, HID * HID);
    cvt(x, XB, NN_NODES * IN_DIM);

    // --- input encoders (lig -> TF, prot -> HF, then per-node select) ---
    k_gemm_bf16<<<gGemm, bGemm, 0, stream>>>(XB, IN_DIM, WB + WB_LIG_W1,  lig_b1,  nullptr, TB, IN_DIM, 1);
    k_gemm_bf16<<<gGemm, bGemm, 0, stream>>>(TB, HID,    WB + WB_LIG_W2,  lig_b2,  TF, nullptr, HID, 1);
    k_gemm_bf16<<<gGemm, bGemm, 0, stream>>>(XB, IN_DIM, WB + WB_PROT_W1, prot_b1, nullptr, TB, IN_DIM, 1);
    k_gemm_bf16<<<gGemm, bGemm, 0, stream>>>(TB, HID,    WB + WB_PROT_W2, prot_b2, HF, nullptr, HID, 1);
    k_select_enc<<<gEl, bEl, 0, stream>>>(x, TF, HF, HB, nEl);

    // --- per-graph node counts ---
    k_zero_f32<<<1, 256, 0, stream>>>(CNT, NGRAPH);
    k_count<<<(NN_NODES + 255) / 256, 256, 0, stream>>>(batch, CNT, NN_NODES);

    for (int l = 0; l < NLAYER; ++l) {
        // GINEConv message passing (aggr in TF)
        k_zero_f32<<<gEl, bEl, 0, stream>>>(TF, nEl);
        k_edge_msg<<<NN_EDGES / 16, 256, 0, stream>>>(ea, src, dst,
            WB + WB_EDGE_W + l * 32 * HID, edge_b + l * HID, HF, TF);
        k_add_cvt<<<gEl, bEl, 0, stream>>>(HF, TF, HB, nEl);   // h = aggr + h
        // nn MLP
        k_gemm_bf16<<<gGemm, bGemm, 0, stream>>>(HB, HID, WB + WB_NN_W1 + l * HID * HID,
                                                 nn_b1 + l * HID, nullptr, TB, HID, 1);
        k_gemm_bf16<<<gGemm, bGemm, 0, stream>>>(TB, HID, WB + WB_NN_W2 + l * HID * HID,
                                                 nn_b2 + l * HID, HF, nullptr, HID, 0);
        // GraphNorm
        k_zero_f32<<<(2 * NGRAPH * HID + 255) / 256, 256, 0, stream>>>(MEAN, 2 * NGRAPH * HID);
        k_gn_sum<<<gEl, bEl, 0, stream>>>(HF, batch, MEAN, nEl);
        k_gn_meandiv<<<(NGRAPH * HID + 255) / 256, 256, 0, stream>>>(MEAN, CNT, NGRAPH * HID);
        k_gn_center<<<gEl, bEl, 0, stream>>>(HF, batch, MEAN, gn_ms + l * HID, VAR, nEl);
        k_gn_norm<<<gEl, bEl, 0, stream>>>(HF, HB, batch, VAR, CNT,
                                           gn_w + l * HID, gn_b + l * HID, nEl);
    }

    // --- global_add_pool + output head ---
    k_zero_f32<<<(NGRAPH * HID + 255) / 256, 256, 0, stream>>>(GP, NGRAPH * HID);
    k_pool<<<gEl, bEl, 0, stream>>>(HF, batch, GP, nEl);
    cvt(GP, GB, NGRAPH * HID);
    k_gemm_bf16<<<dim3(NGRAPH / 16), bGemm, 0, stream>>>(GB, HID, WB + WB_FC_W1,
                                                         fc_b1, HEAD, nullptr, HID, 1);
    k_head<<<1, NGRAPH, 0, stream>>>(HEAD, fc_w2, fc_b2, (float*)d_out);
}
